// VectorQuantizer_5798205850204
// MI455X (gfx1250) — matrix operand order repaired and checked by reading, not compiled, not run on hardware
//
#include <hip/hip_runtime.h>

#define NUM_EMB        8192
#define EMB_DIM        256
#define N_ROWS         32768        // 64 * 512
#define ROWS_PER_BLOCK 128
#define CODES_PER_ITER 32           // two 16-column WMMA sub-tiles per stage
#define NUM_ITERS      (NUM_EMB / CODES_PER_ITER)   // 256
#define STAGE_BYTES    16384        // 32 codes * 256 * 2B
#define GATHER_BLOCKS  4096
#define TOTAL_ELEMS    8388608      // N_ROWS * EMB_DIM

typedef __attribute__((ext_vector_type(16))) __bf16 v16bf;
typedef __attribute__((ext_vector_type(8)))  float  v8f;

union BFrag {
    uint4          u[2];
    unsigned short s[16];
    v16bf          v;
};

static __device__ __forceinline__ unsigned short f32_to_bf16_rne(float f) {
    unsigned int u = __float_as_uint(f);
    u += 0x7FFFu + ((u >> 16) & 1u);
    return (unsigned short)(u >> 16);
}

static __device__ __forceinline__ float clip1(float v) {
    return fminf(fmaxf(v, -1.0f), 1.0f);
}

// Fire-and-forget 16KB stage: 256 threads x 4 x b128 async copies to LDS
// (ASYNCcnt). No "memory" clobber: ordering vs. the wait is guaranteed by
// volatile-asm program order, and __syncthreads() fences buffer reuse --
// this keeps the instruction scheduler free to pipeline the WMMA loop.
static __device__ __forceinline__ void stage_async(const unsigned short* ebf,
                                                   int iter, unsigned ldsDst) {
    unsigned long long g = (unsigned long long)(uintptr_t)
        (reinterpret_cast<const char*>(ebf) + (size_t)iter * STAGE_BYTES +
         (size_t)threadIdx.x * 16u);
    unsigned l = ldsDst + threadIdx.x * 16u;
    #pragma unroll
    for (int p = 0; p < 4; ++p) {
        asm volatile("global_load_async_to_lds_b128 %0, %1, off"
                     :: "v"(l + (unsigned)p * 4096u),
                        "v"(g + (unsigned long long)p * 4096ull));
    }
}

static __device__ __forceinline__ void wait_async0() {
    asm volatile("s_wait_asynccnt 0x0" ::: "memory");
}

// ---------------------------------------------------------------------------
// Kernel 1: codebook -> bf16, and h[k] = 0.5 * ||e_k||^2
// ---------------------------------------------------------------------------
__global__ __launch_bounds__(256)
void vq_prep_kernel(const float* __restrict__ emb,
                    unsigned short* __restrict__ ebf,
                    float* __restrict__ hn) {
    __shared__ float red[256];
    const int k = blockIdx.x;
    const int d = threadIdx.x;
    const float e = emb[(size_t)k * EMB_DIM + d];
    ebf[(size_t)k * EMB_DIM + d] = f32_to_bf16_rne(e);
    red[d] = e * e;
    __syncthreads();
    for (int s = 128; s > 0; s >>= 1) {
        if (d < s) red[d] += red[d + s];
        __syncthreads();
    }
    if (d == 0) hn[k] = 0.5f * red[0];
}

// ---------------------------------------------------------------------------
// Kernel 2: bf16 WMMA GEMM + running argmax of (x . e_k - 0.5*||e_k||^2)
//   block = 256 threads = 8 waves; wave w owns rows blockIdx*128 + 16*w
//   32 codes per iteration, double-buffered in LDS via async-to-LDS copies;
//   the two 16-col sub-tiles are interleaved on two accumulators.
// ---------------------------------------------------------------------------
__global__ __launch_bounds__(256)
void vq_argmin_kernel(const float* __restrict__ x,
                      const unsigned short* __restrict__ ebf,
                      const float* __restrict__ hn,
                      int* __restrict__ idxout) {
    __shared__ uint4 sB[2][1024];   // 2 x 16KB stages (32 codes x 256 bf16)
    const unsigned ldsBase = (unsigned)(uintptr_t)(&sB[0][0]);  // LDS uses addr[31:0]

    const int lane    = threadIdx.x & 31;
    const int wave    = threadIdx.x >> 5;
    const int half    = lane >> 4;     // 0 or 1 (half-wave)
    const int l15     = lane & 15;
    const int rowbase = blockIdx.x * ROWS_PER_BLOCK + wave * 16;

    // kick off stage 0 before doing the A-fragment conversion work
    stage_async(ebf, 0, ldsBase);

    // --- Load A fragments once (ISA 16-bit A 16x32 layout), keep in VGPRs ---
    BFrag afrag[8];
    {
        const float* xrow = x + (size_t)(rowbase + l15) * EMB_DIM;
        #pragma unroll
        for (int kk = 0; kk < 8; ++kk) {
            const int d0 = kk * 32 + half * 8;   // K {0..7|8..15} by half-wave
            float va[8], vb[8];
            *(float4*)&va[0] = *reinterpret_cast<const float4*>(xrow + d0);
            *(float4*)&va[4] = *reinterpret_cast<const float4*>(xrow + d0 + 4);
            *(float4*)&vb[0] = *reinterpret_cast<const float4*>(xrow + d0 + 16);
            *(float4*)&vb[4] = *reinterpret_cast<const float4*>(xrow + d0 + 20);
            #pragma unroll
            for (int i = 0; i < 8; ++i) {
                afrag[kk].s[i]     = f32_to_bf16_rne(clip1(va[i]));
                afrag[kk].s[8 + i] = f32_to_bf16_rne(clip1(vb[i]));
            }
        }
    }

    float best0[8], best1[8];
    int   bidx0[8], bidx1[8];
    #pragma unroll
    for (int j = 0; j < 8; ++j) {
        best0[j] = -3.402823466e38f; bidx0[j] = 0x7fffffff;
        best1[j] = -3.402823466e38f; bidx1[j] = 0x7fffffff;
    }

    wait_async0();
    __syncthreads();

    for (int t = 0; t < NUM_ITERS; ++t) {
        const int cur = t & 1;
        if (t + 1 < NUM_ITERS) {
            stage_async(ebf, t + 1, ldsBase + (unsigned)(cur ^ 1) * STAGE_BYTES);
            if (t + 2 < NUM_ITERS)
                __builtin_prefetch(reinterpret_cast<const char*>(ebf) +
                                   (size_t)(t + 2) * STAGE_BYTES +
                                   (size_t)threadIdx.x * 64, 0, 1);
        }

        const int   code0 = t * CODES_PER_ITER + l15;
        const int   code1 = code0 + 16;
        const float h0 = hn[code0];
        const float h1 = hn[code1];

        const uint4* sbuf = &sB[cur][0];
        v8f c0 = {}, c1 = {};
        #pragma unroll
        for (int kk = 0; kk < 8; ++kk) {
            // B 32x16: lanes 0-15 = K {0..7,16..23} (col=lane), lanes 16-31 = K {8..15,24..31}
            BFrag b0, b1;
            const int o = l15 * 32 + kk * 4 + half;  // uint4 index in sub-tile
            b0.u[0] = sbuf[o];        b0.u[1] = sbuf[o + 2];
            b1.u[0] = sbuf[o + 512];  b1.u[1] = sbuf[o + 514];
            c0 = __builtin_amdgcn_wmma_f32_16x16x32_bf16(
                     false, afrag[kk].v, false, b0.v, (short)0, c0, false, false);
            c1 = __builtin_amdgcn_wmma_f32_16x16x32_bf16(
                     false, afrag[kk].v, false, b1.v, (short)0, c1, false, false);
        }

        // strict '>' keeps the first (lowest) index on ties: code is strictly
        // increasing within a lane, matching jnp.argmin order.
        #pragma unroll
        for (int j = 0; j < 8; ++j) {
            const float s0 = c0[j] - h0;
            if (s0 > best0[j]) { best0[j] = s0; bidx0[j] = code0; }
            const float s1 = c1[j] - h1;
            if (s1 > best1[j]) { best1[j] = s1; bidx1[j] = code1; }
        }

        wait_async0();     // staged data for t+1 resident in LDS
        __syncthreads();   // all waves done reading buffer `cur`
    }

    // merge the two column groups (code0 < code1 always -> '>' keeps lower idx)
    float best[8]; int bidx[8];
    #pragma unroll
    for (int j = 0; j < 8; ++j) {
        best[j] = best0[j]; bidx[j] = bidx0[j];
        if (best1[j] > best[j]) { best[j] = best1[j]; bidx[j] = bidx1[j]; }
    }

    // butterfly argmax across the 16 columns held by each half-wave
    // (tie-break on lowest index, matching jnp.argmin)
    #pragma unroll
    for (int off = 1; off < 16; off <<= 1) {
        #pragma unroll
        for (int j = 0; j < 8; ++j) {
            const float ov = __shfl_xor(best[j], off, 16);
            const int   oi = __shfl_xor(bidx[j], off, 16);
            if (ov > best[j] || (ov == best[j] && oi < bidx[j])) {
                best[j] = ov; bidx[j] = oi;
            }
        }
    }
    if (l15 == 0) {
        #pragma unroll
        for (int j = 0; j < 8; ++j)
            idxout[rowbase + half * 8 + j] = bidx[j];   // C/D: row = j + 8*half
    }
}

// ---------------------------------------------------------------------------
// Kernel 3: fp32 gather of winning codes + per-block squared-error partials
// ---------------------------------------------------------------------------
__global__ __launch_bounds__(256)
void vq_gather_kernel(const float* __restrict__ x,
                      const float* __restrict__ emb,
                      const int* __restrict__ idx,
                      float* __restrict__ out,
                      float* __restrict__ partial) {
    __shared__ float red[256];
    float acc = 0.f;
    const int base = blockIdx.x * 2048;
    #pragma unroll
    for (int i = 0; i < 8; ++i) {
        const int e = base + i * 256 + threadIdx.x;
        const int n = e >> 8;
        const int d = e & 255;
        const float q  = emb[(size_t)idx[n] * EMB_DIM + d];
        const float xv = clip1(x[e]);
        out[e] = q;                  // x + stop_grad(q - x) == q numerically
        const float df = q - xv;
        acc += df * df;
    }
    red[threadIdx.x] = acc;
    __syncthreads();
    for (int s = 128; s > 0; s >>= 1) {
        if (threadIdx.x < s) red[threadIdx.x] += red[threadIdx.x + s];
        __syncthreads();
    }
    if (threadIdx.x == 0) partial[blockIdx.x] = red[0];
}

// ---------------------------------------------------------------------------
// Kernel 4: final reduction -> loss = 1.25 * mean((q - x)^2)
// ---------------------------------------------------------------------------
__global__ __launch_bounds__(256)
void vq_loss_kernel(const float* __restrict__ partial, float* __restrict__ loss_out) {
    __shared__ double red[256];
    double a = 0.0;
    for (int i = threadIdx.x; i < GATHER_BLOCKS; i += 256) a += (double)partial[i];
    red[threadIdx.x] = a;
    __syncthreads();
    for (int s = 128; s > 0; s >>= 1) {
        if (threadIdx.x < s) red[threadIdx.x] += red[threadIdx.x + s];
        __syncthreads();
    }
    if (threadIdx.x == 0)
        loss_out[0] = (float)(1.25 * red[0] / (double)TOTAL_ELEMS);
}

// ---------------------------------------------------------------------------
extern "C" void kernel_launch(void* const* d_in, const int* in_sizes, int n_in,
                              void* d_out, int out_size, void* d_ws, size_t ws_size,
                              hipStream_t stream) {
    (void)in_sizes; (void)n_in; (void)out_size; (void)ws_size;

    const float* x   = (const float*)d_in[0];   // [64,512,256] f32
    const float* emb = (const float*)d_in[1];   // [8192,256]  f32
    float* out = (float*)d_out;                 // 8388608 quantized + 1 loss

    char* ws = (char*)d_ws;
    unsigned short* ebf     = (unsigned short*)(ws);                           // 4 MB
    float*          hn      = (float*)(ws + 4u * 1024u * 1024u);               // 32 KB
    int*            idx     = (int*)  (ws + 4u * 1024u * 1024u + 64u * 1024u); // 128 KB
    float*          partial = (float*)(ws + 4u * 1024u * 1024u + 320u * 1024u);// 16 KB

    vq_prep_kernel  <<<NUM_EMB, 256, 0, stream>>>(emb, ebf, hn);
    vq_argmin_kernel<<<N_ROWS / ROWS_PER_BLOCK, 256, 0, stream>>>(x, ebf, hn, idx);
    vq_gather_kernel<<<GATHER_BLOCKS, 256, 0, stream>>>(x, emb, idx, out, partial);
    vq_loss_kernel  <<<1, 256, 0, stream>>>(partial, out + TOTAL_ELEMS);
}
